// CPCLoss_51694226375287
// MI455X (gfx1250) — matrix operand order repaired and checked by reading, not compiled, not run on hardware
//
#include <hip/hip_runtime.h>
#include <hip/hip_bf16.h>
#include <math.h>

typedef __attribute__((ext_vector_type(16))) __bf16 v16bf;
typedef __attribute__((ext_vector_type(8)))  __bf16 v8bf;
typedef __attribute__((ext_vector_type(4)))  __bf16 v4bf;
typedef __attribute__((ext_vector_type(8)))  float  v8f;

#define N_SPK 8
#define N_UTT 8
#define N_SU  64
#define N_K   12
#define N_NEG 17
#define ZD    64
#define CD    256
#define T_DIM 1140
#define LEN   1128            // T - K
#define LTILES 71             // ceil(1128/16), last tile partial (8 valid rows)
#define SCALE 0.125f          // 1/sqrt(ZD)

// round-to-nearest-even f32 -> bf16
__device__ __forceinline__ __bf16 cvt_bf16(float f) {
  union { float f; unsigned u; } in; in.f = f;
  unsigned u = in.u;
  u += 0x7FFFu + ((u >> 16) & 1u);
  unsigned short h = (unsigned short)(u >> 16);
  union { unsigned short s; __bf16 b; } out; out.s = h;
  return out.b;
}

__global__ void f32_to_bf16_vec(const float* __restrict__ src,
                                __bf16* __restrict__ dst, int n4) {
  int i = blockIdx.x * blockDim.x + threadIdx.x;
  if (i < n4) {
    const float4 v = ((const float4*)src)[i];
    v4bf o;
    o[0] = cvt_bf16(v.x); o[1] = cvt_bf16(v.y);
    o[2] = cvt_bf16(v.z); o[3] = cvt_bf16(v.w);
    *(v4bf*)(dst + 4 * (size_t)i) = o;
  }
}

__global__ void zero_out(float* out) {
  if (threadIdx.x < 1 + N_K) out[threadIdx.x] = 0.f;
}

// grid: (l-tile 0..70, su 0..63, k 0..11); block: 128 threads = 4 waves
template <bool PRECONV>
__launch_bounds__(128)
__global__ void cpc_fused(const float* __restrict__ z,
                          const float* __restrict__ c,
                          const __bf16* __restrict__ cbf,
                          const __bf16* __restrict__ Wbf,
                          const float* __restrict__ b,
                          const int* __restrict__ batch_index,
                          const int* __restrict__ seq_index,
                          float* __restrict__ out)
{
  __shared__ float lds_wc[16 * ZD];           // scaled Wc tile (16 l-rows x 64 z)
  __shared__ float lds_f[16 * (N_NEG + 1)];   // scores
  __shared__ float lds_red[32];

  const int tid  = threadIdx.x;
  const int lane = tid & 31;
  const int wave = tid >> 5;
  const int half = lane >> 4;
  const int lo16 = lane & 15;

  const int ltile = blockIdx.x;
  const int su    = blockIdx.y;
  const int kidx  = blockIdx.z;               // reference k = kidx + 1
  const int shift = kidx + 1;
  const int s     = su >> 3;
  const int u     = su & 7;
  const int l0    = ltile * 16;

  // ---------- Phase 1: Wc tile = c_tile @ W[k]^T via bf16 WMMA (f32 acc) ----------
  // This wave owns output columns z in [n0, n0+16). B[K=d][N=z] = W[k][z][d].
  const int n0   = wave * 16;
  const int zcol = n0 + lo16;

  // B fragments: 32x16 16-bit layout -> lane = N column; lanes 0-15 hold K {0..7, 16..23},
  // lanes 16-31 hold K {8..15, 24..31} => two 16B-aligned v8bf loads per chunk.
  const __bf16* wrow = Wbf + ((size_t)kidx * ZD + zcol) * CD;
  v16bf Bfrag[8];
#pragma unroll
  for (int kk = 0; kk < 8; ++kk) {
    const v8bf blo = *(const v8bf*)(wrow + kk * 32 + half * 8);
    const v8bf bhi = *(const v8bf*)(wrow + kk * 32 + half * 8 + 16);
    Bfrag[kk] = __builtin_shufflevector(blo, bhi,
        0, 1, 2, 3, 4, 5, 6, 7, 8, 9, 10, 11, 12, 13, 14, 15);
  }

  // A row for this lane (clamped on the partial last tile; masked in phase 2)
  int lrow = l0 + lo16; if (lrow > LEN - 1) lrow = LEN - 1;

  v8f acc = {0.f, 0.f, 0.f, 0.f, 0.f, 0.f, 0.f, 0.f};

  // 16-bit A 16x32 layout: lanes 0-15 hold K {0..7, 16..23}, lanes 16-31 hold
  // K {8..15, 24..31} -> two 8-element contiguous runs per lane per chunk.
  if (PRECONV) {
    const __bf16* crow = cbf + ((size_t)su * T_DIM + lrow) * CD;
#pragma unroll
    for (int kk = 0; kk < 8; ++kk) {
      const int dbase = kk * 32 + half * 8;
      const v8bf lo = *(const v8bf*)(crow + dbase);        // 16B aligned b128
      const v8bf hi = *(const v8bf*)(crow + dbase + 16);   // 16B aligned b128
      const v16bf a = __builtin_shufflevector(lo, hi,
          0, 1, 2, 3, 4, 5, 6, 7, 8, 9, 10, 11, 12, 13, 14, 15);
      acc = __builtin_amdgcn_wmma_f32_16x16x32_bf16(false, a, false, Bfrag[kk],
                                                    (short)0, acc, false, false);
    }
  } else {
    const float* crow = c + ((size_t)su * T_DIM + lrow) * CD;
#pragma unroll
    for (int kk = 0; kk < 8; ++kk) {
      const int dbase = kk * 32 + half * 8;
      const float4 r0 = *(const float4*)(crow + dbase);
      const float4 r1 = *(const float4*)(crow + dbase + 4);
      const float4 r2 = *(const float4*)(crow + dbase + 16);
      const float4 r3 = *(const float4*)(crow + dbase + 20);
      v16bf a;
      a[0]  = cvt_bf16(r0.x); a[1]  = cvt_bf16(r0.y); a[2]  = cvt_bf16(r0.z); a[3]  = cvt_bf16(r0.w);
      a[4]  = cvt_bf16(r1.x); a[5]  = cvt_bf16(r1.y); a[6]  = cvt_bf16(r1.z); a[7]  = cvt_bf16(r1.w);
      a[8]  = cvt_bf16(r2.x); a[9]  = cvt_bf16(r2.y); a[10] = cvt_bf16(r2.z); a[11] = cvt_bf16(r2.w);
      a[12] = cvt_bf16(r3.x); a[13] = cvt_bf16(r3.y); a[14] = cvt_bf16(r3.z); a[15] = cvt_bf16(r3.w);
      acc = __builtin_amdgcn_wmma_f32_16x16x32_bf16(false, a, false, Bfrag[kk],
                                                    (short)0, acc, false, false);
    }
  }

  // C/D layout: VGPR v -> M = half*8 + v, N = lo16. Fold +b and the 1/sqrt(ZD)
  // scale into Wc (both pos and neg dots multiply the same Wc).
  const float bval = b[kidx * ZD + zcol];
#pragma unroll
  for (int v = 0; v < 8; ++v)
    lds_wc[(half * 8 + v) * ZD + zcol] = (acc[v] + bval) * SCALE;
  __syncthreads();

  // ---------- Phase 2: pos + 17 gathered neg dots, logsumexp, accumulate ----------
  for (int task = tid; task < 16 * (N_NEG + 1); task += 128) {
    const int l  = task / (N_NEG + 1);
    const int n  = task - l * (N_NEG + 1);
    const int lg = l0 + l;
    if (lg < LEN) {
      const float* zrow;
      if (n == 0) {
        zrow = z + ((size_t)su * T_DIM + (lg + shift)) * ZD;
      } else {
        const int nn = n - 1;
        const int bi = batch_index[(kidx * N_UTT + u) * N_NEG + nn];
        const size_t sioff =
            ((((size_t)kidx * N_SPK + s) * N_UTT + u) * N_NEG + nn) * (size_t)LEN + lg;
        const int si = seq_index[sioff];
        zrow = z + ((size_t)(s * N_UTT + bi) * T_DIM + (si + shift)) * ZD;
      }
      const float4* zp = (const float4*)zrow;
      const float4* wp = (const float4*)(lds_wc + l * ZD);
      float dot = 0.f;
#pragma unroll
      for (int j = 0; j < 16; ++j) {
        const float4 zv = zp[j];
        const float4 wv = wp[j];
        dot = fmaf(zv.x, wv.x, dot);
        dot = fmaf(zv.y, wv.y, dot);
        dot = fmaf(zv.z, wv.z, dot);
        dot = fmaf(zv.w, wv.w, dot);
      }
      lds_f[task] = dot;   // already scaled via Wc
    }
  }
  __syncthreads();

  if (tid < 16) {
    float lossv = 0.f, accv = 0.f;
    if (l0 + tid < LEN) {
      const float* f = lds_f + tid * (N_NEG + 1);
      float m = f[0];
#pragma unroll
      for (int i = 1; i < N_NEG + 1; ++i) m = fmaxf(m, f[i]);
      float ssum = 0.f;
#pragma unroll
      for (int i = 0; i < N_NEG + 1; ++i) ssum += __expf(f[i] - m);
      const float logz = m + __logf(ssum);
      lossv = logz - f[0];
      accv  = (f[0] >= m) ? 1.f : 0.f;   // argmax == 0 (ties resolve to 0)
    }
    lds_red[tid]      = lossv;
    lds_red[16 + tid] = accv;
  }
  __syncthreads();
  if (tid == 0) {
    float ls = 0.f, ac = 0.f;
#pragma unroll
    for (int i = 0; i < 16; ++i) { ls += lds_red[i]; ac += lds_red[16 + i]; }
    atomicAdd(&out[0], ls);          // summed over all k; normalized in finalize
    atomicAdd(&out[1 + kidx], ac);
  }
}

__global__ void finalize(float* out) {
  const float denom = (float)N_SU * (float)LEN;   // 72192 elements per k
  const int t = threadIdx.x;
  if (t == 0)          out[0] = out[0] / (denom * (float)N_K);
  else if (t <= N_K)   out[t] = out[t] / denom;
}

extern "C" void kernel_launch(void* const* d_in, const int* in_sizes, int n_in,
                              void* d_out, int out_size, void* d_ws, size_t ws_size,
                              hipStream_t stream) {
  const float* z           = (const float*)d_in[0];
  const float* c           = (const float*)d_in[1];
  const float* W           = (const float*)d_in[2];
  const float* b           = (const float*)d_in[3];
  const int*   batch_index = (const int*)d_in[4];
  const int*   seq_index   = (const int*)d_in[5];
  float* out = (float*)d_out;

  const size_t wN  = (size_t)N_K * ZD * CD;          // 196,608
  const size_t cN  = (size_t)N_SU * T_DIM * CD;      // 18,677,760
  const size_t wsNeeded = (wN + cN) * sizeof(__bf16);// ~37.75 MB

  __bf16* Wbf = (__bf16*)d_ws;                       // [0, 384 KB)
  __bf16* cbf = Wbf + wN;                            // [384 KB, ~37.75 MB)

  // W -> bf16 (always; only 384 KB of ws)
  f32_to_bf16_vec<<<(int)((wN / 4 + 255) / 256), 256, 0, stream>>>(W, Wbf, (int)(wN / 4));
  zero_out<<<1, 32, 0, stream>>>(out);

  dim3 grid(LTILES, N_SU, N_K);
  if (ws_size >= wsNeeded) {
    // c -> bf16 once (kills all per-block conversion VALU; A loads become b128)
    f32_to_bf16_vec<<<(int)((cN / 4 + 255) / 256), 256, 0, stream>>>(c, cbf, (int)(cN / 4));
    cpc_fused<true><<<grid, 128, 0, stream>>>(z, c, cbf, Wbf, b,
                                              batch_index, seq_index, out);
  } else {
    cpc_fused<false><<<grid, 128, 0, stream>>>(z, c, cbf, Wbf, b,
                                               batch_index, seq_index, out);
  }

  finalize<<<1, 32, 0, stream>>>(out);
}
